// CMlp_75565654606266
// MI455X (gfx1250) — matrix operand order repaired and checked by reading, not compiled, not run on hardware
//
#include <hip/hip_runtime.h>
#include <math.h>

// ---------------------------------------------------------------------------
// CMlp forward for MI455X (gfx1250): PLIF -> 1x1conv(GEMM) -> PLIF -> 1x1conv
// f16 WMMA (16x16x32, f32 accum); spikes are exactly representable in f16.
// ---------------------------------------------------------------------------

typedef _Float16 v16h __attribute__((ext_vector_type(16)));
typedef _Float16 v8h  __attribute__((ext_vector_type(8)));
typedef float    v8f  __attribute__((ext_vector_type(8)));

#define TT   4
#define BB   32
#define CC   384
#define HID  1536
#define HWW  196
#define NPT  (BB * HWW)   // 6272 positions per timestep
#define NTOT (TT * NPT)   // 25088 total positions

static __device__ __forceinline__ v16h cat16(v8h lo, v8h hi) {
  return __builtin_shufflevector(lo, hi, 0,1,2,3,4,5,6,7,8,9,10,11,12,13,14,15);
}

// A-matrix fragment (16x32 f16, M x K). Per ISA layout: lanes 0-15 hold row
// M=lane, K {0..7} in VGPRs0-3 and {16..23} in VGPRs4-7; lanes 16-31 hold
// K {8..15} and {24..31}. Two aligned 16B loads per lane.
static __device__ __forceinline__ v16h load_fragA(const _Float16* __restrict__ base,
                                                  int ld, int m, int k0, int lane) {
  const int hi8 = (lane >= 16) ? 8 : 0;
  const _Float16* row = base + (size_t)m * ld + k0 + hi8;
  v8h lo = *(const v8h*)(row);
  v8h hi = *(const v8h*)(row + 16);
  return cat16(lo, hi);
}

// B-matrix fragment (32x16 f16, K x N). Lane holds column N=lane&15; lanes
// 0-15 cover K {0..7},{16..23}, lanes 16-31 cover K {8..15},{24..31}.
static __device__ __forceinline__ v16h load_fragB(const _Float16* __restrict__ base,
                                                  int ld, int n, int k0, int lane) {
  const int kk = k0 + ((lane >= 16) ? 8 : 0);
  const v8h* p = (const v8h*)(base + (size_t)n * ld + kk);
  return cat16(p[0], p[2]);
}

// ---------------------------------------------------------------------------
// Kernel 0: convert fp32 weights to f16 once (they live in L2 afterwards).
// ---------------------------------------------------------------------------
__global__ __launch_bounds__(256) void cvt_weights(const float* __restrict__ w1,
                                                   const float* __restrict__ w2,
                                                   _Float16* __restrict__ w1h,
                                                   _Float16* __restrict__ w2h) {
  const int n = HID * CC;
  int i = blockIdx.x * 256 + threadIdx.x;
  if (i < n) {
    w1h[i] = (_Float16)w1[i];
  } else {
    int j = i - n;
    w2h[j] = (_Float16)w2[j];
  }
}

// ---------------------------------------------------------------------------
// Kernel 1: PLIF over x. T-recurrence in registers (T=4). Emits binary spikes
// as f16 in layout s1[t][n][c] (c contiguous => K-contiguous B fragments).
// ---------------------------------------------------------------------------
__global__ __launch_bounds__(256) void plif1_kernel(const float* __restrict__ x,
                                                    const float* __restrict__ pw1,
                                                    _Float16* __restrict__ s1) {
  int idx = blockIdx.x * 256 + threadIdx.x;      // over B*C*HW (exact)
  int b   = idx / (CC * HWW);
  int rem = idx % (CC * HWW);
  int c   = rem / HWW;
  int hw  = rem % HWW;
  int n   = b * HWW + hw;
  const float decay = 1.0f / (1.0f + __expf(-pw1[0]));
  float v = 0.0f;
#pragma unroll
  for (int t = 0; t < TT; ++t) {
    float xv = x[(((size_t)t * BB + b) * CC + c) * HWW + hw];
    float h  = v + (xv - v) * decay;
    bool fire = (h - 1.0f) >= 0.0f;
    v = fire ? 0.0f : h;                          // hard reset to 0
    s1[((size_t)t * NPT + n) * CC + c] = (_Float16)(fire ? 1.0f : 0.0f);
  }
}

// One PLIF step on an 8-element accumulator strip; packs spikes to v8h.
static __device__ __forceinline__ void plif_step(v8f acc, const float* __restrict__ bias,
                                                 float* __restrict__ v, float decay,
                                                 _Float16* __restrict__ dst) {
  v8h pk;
#pragma unroll
  for (int i = 0; i < 8; ++i) {
    float h = v[i] + ((acc[i] + bias[i]) - v[i]) * decay;
    bool fire = (h - 1.0f) >= 0.0f;
    v[i] = fire ? 0.0f : h;
    pk[i] = (_Float16)(fire ? 1.0f : 0.0f);
  }
  *(v8h*)dst = pk;                                // 16B aligned store
}

// ---------------------------------------------------------------------------
// Kernel 2: GEMM1 (h = s1 @ w1^T + b1) fused with PLIF2.
// Block = 8 waves = 32(M) x 64(N) tile; each wave keeps FOUR accumulators
// (one per timestep) sharing the same A (weight) fragment, so the t-recurrence
// runs entirely in registers after the K loop.
// ---------------------------------------------------------------------------
__global__ __launch_bounds__(256) void gemm1_plif2(const _Float16* __restrict__ w1h,
                                                   const float* __restrict__ b1,
                                                   const _Float16* __restrict__ s1,
                                                   const float* __restrict__ pw2,
                                                   _Float16* __restrict__ s2) {
  const int lane = threadIdx.x & 31;
  const int wave = threadIdx.x >> 5;
  const int m0 = blockIdx.x * 32 + (wave >> 2) * 16;   // hid rows
  const int n0 = blockIdx.y * 64 + (wave & 3) * 16;    // positions
  const int mA = m0 + (lane & 15);
  const int nB = n0 + (lane & 15);

  const _Float16* s1t0 = s1 + 0 * (size_t)NPT * CC;
  const _Float16* s1t1 = s1 + 1 * (size_t)NPT * CC;
  const _Float16* s1t2 = s1 + 2 * (size_t)NPT * CC;
  const _Float16* s1t3 = s1 + 3 * (size_t)NPT * CC;

  v8f acc0 = {}, acc1 = {}, acc2 = {}, acc3 = {};
  for (int k0 = 0; k0 < CC; k0 += 32) {
    if (k0 + 32 < CC) {  // prefetch next K-chunk of the streaming operands
      __builtin_prefetch(s1t0 + (size_t)nB * CC + k0 + 32, 0, 1);
      __builtin_prefetch(s1t3 + (size_t)nB * CC + k0 + 32, 0, 1);
    }
    v16h a  = load_fragA(w1h, CC, mA, k0, lane);
    v16h b0 = load_fragB(s1t0, CC, nB, k0, lane);
    acc0 = __builtin_amdgcn_wmma_f32_16x16x32_f16(false, a, false, b0, (short)0, acc0, false, false);
    v16h b1f = load_fragB(s1t1, CC, nB, k0, lane);
    acc1 = __builtin_amdgcn_wmma_f32_16x16x32_f16(false, a, false, b1f, (short)0, acc1, false, false);
    v16h b2f = load_fragB(s1t2, CC, nB, k0, lane);
    acc2 = __builtin_amdgcn_wmma_f32_16x16x32_f16(false, a, false, b2f, (short)0, acc2, false, false);
    v16h b3f = load_fragB(s1t3, CC, nB, k0, lane);
    acc3 = __builtin_amdgcn_wmma_f32_16x16x32_f16(false, a, false, b3f, (short)0, acc3, false, false);
  }

  // C/D layout: VGPR i, lanes 0-15 -> M=i, lanes 16-31 -> M=8+i; N=lane&15.
  const int ostart = m0 + (lane >> 4) * 8;
  float bias[8];
#pragma unroll
  for (int i = 0; i < 8; ++i) bias[i] = b1[ostart + i];

  const float decay = 1.0f / (1.0f + __expf(-pw2[0]));
  float v[8];
#pragma unroll
  for (int i = 0; i < 8; ++i) v[i] = 0.0f;

  _Float16* dbase = s2 + (size_t)nB * HID + ostart;
  plif_step(acc0, bias, v, decay, dbase + 0 * (size_t)NPT * HID);
  plif_step(acc1, bias, v, decay, dbase + 1 * (size_t)NPT * HID);
  plif_step(acc2, bias, v, decay, dbase + 2 * (size_t)NPT * HID);
  plif_step(acc3, bias, v, decay, dbase + 3 * (size_t)NPT * HID);
}

// ---------------------------------------------------------------------------
// Kernel 3: GEMM2 (out = s2 @ w2^T + b2), t folded into N (25088 columns),
// scatter f32 results into [T,B,O,H,W] layout.
// ---------------------------------------------------------------------------
__global__ __launch_bounds__(256) void gemm2_kernel(const _Float16* __restrict__ w2h,
                                                    const float* __restrict__ b2,
                                                    const _Float16* __restrict__ s2,
                                                    float* __restrict__ out) {
  const int lane = threadIdx.x & 31;
  const int wave = threadIdx.x >> 5;
  const int m0 = blockIdx.x * 32 + (wave >> 2) * 16;   // output channels
  const int n0 = blockIdx.y * 64 + (wave & 3) * 16;    // flattened t*b*hw
  const int mA = m0 + (lane & 15);
  const int nB = n0 + (lane & 15);

  v8f acc = {};
  for (int k0 = 0; k0 < HID; k0 += 32) {
    if (k0 + 32 < HID)
      __builtin_prefetch(s2 + (size_t)nB * HID + k0 + 32, 0, 1);
    v16h a = load_fragA(w2h, HID, mA, k0, lane);
    v16h b = load_fragB(s2, HID, nB, k0, lane);
    acc = __builtin_amdgcn_wmma_f32_16x16x32_f16(false, a, false, b, (short)0, acc, false, false);
  }

  const int t  = nB / NPT;
  const int r  = nB % NPT;
  const int bb = r / HWW;
  const int hw = r % HWW;
  const int ostart = m0 + (lane >> 4) * 8;
#pragma unroll
  for (int i = 0; i < 8; ++i) {
    int o = ostart + i;
    out[(((size_t)t * BB + bb) * CC + o) * HWW + hw] = acc[i] + b2[o];
  }
}

// ---------------------------------------------------------------------------
// Host-side launcher.
// Inputs (setup_inputs order): x, pw1, w1, b1, pw2, w2, b2   (all fp32)
// Workspace layout: w1h | w2h | s1[t][n][c] f16 | s2[t][n][o] f16  (~94 MB)
// ---------------------------------------------------------------------------
extern "C" void kernel_launch(void* const* d_in, const int* in_sizes, int n_in,
                              void* d_out, int out_size, void* d_ws, size_t ws_size,
                              hipStream_t stream) {
  (void)in_sizes; (void)n_in; (void)out_size; (void)ws_size;
  const float* x   = (const float*)d_in[0];
  const float* pw1 = (const float*)d_in[1];
  const float* w1  = (const float*)d_in[2];
  const float* b1  = (const float*)d_in[3];
  const float* pw2 = (const float*)d_in[4];
  const float* w2  = (const float*)d_in[5];
  const float* b2  = (const float*)d_in[6];
  float* out = (float*)d_out;

  _Float16* w1h = (_Float16*)d_ws;
  _Float16* w2h = w1h + (size_t)HID * CC;
  _Float16* s1  = w2h + (size_t)HID * CC;
  _Float16* s2  = s1 + (size_t)TT * NPT * CC;

  cvt_weights<<<(2 * HID * CC) / 256, 256, 0, stream>>>(w1, w2, w1h, w2h);
  plif1_kernel<<<(BB * CC * HWW) / 256, 256, 0, stream>>>(x, pw1, s1);
  gemm1_plif2<<<dim3(HID / 32, NPT / 64), 256, 0, stream>>>(w1h, b1, s1, pw2, s2);
  gemm2_kernel<<<dim3(CC / 32, NTOT / 64), 256, 0, stream>>>(w2h, b2, s2, out);
}
